// CausalSelfAttention_7533372637546
// MI455X (gfx1250) — matrix operand order repaired and checked by reading, not compiled, not run on hardware
//
#include <hip/hip_runtime.h>
#include <hip/hip_bf16.h>

// ---------------------------------------------------------------------------
// Causal self-attention (B=4, T=2048, C=768, H=12, Dh=64) for gfx1250.
// - All matmuls: v_wmma_f32_16x16x32_bf16 (fp32 accumulate).
// - GEMMs: 128x64 block tile; block-shared B slabs double-buffered in LDS via
//   GLOBAL_LOAD_ASYNC_TO_LDS_B128 (ASYNCcnt), A streamed with global b128.
// - Attention: S^T = K@Q^T trick keeps softmax reductions intra-lane.
// ---------------------------------------------------------------------------

typedef __bf16 bf16;
typedef __attribute__((ext_vector_type(16))) __bf16 bf16x16;
typedef __attribute__((ext_vector_type(8)))  __bf16 bf16x8;
typedef __attribute__((ext_vector_type(8)))  float   f32x8;
typedef __attribute__((ext_vector_type(4)))  int     i32x4;

namespace {
constexpr int kB  = 4;
constexpr int kT  = 2048;
constexpr int kC  = 768;
constexpr int kH  = 12;
constexpr int kD  = 64;
constexpr int kM  = kB * kT;   // 8192 rows of x
constexpr int kN3 = 3 * kC;    // 2304
}

#if __has_builtin(__builtin_amdgcn_global_load_async_to_lds_b128) && \
    __has_builtin(__builtin_amdgcn_s_wait_asynccnt)
#define USE_ASYNC_LDS 1
#else
#define USE_ASYNC_LDS 0
#endif

#define AS1 __attribute__((address_space(1)))
#define AS3 __attribute__((address_space(3)))

__device__ __forceinline__ f32x8 wmma_bf16(bf16x16 a, bf16x16 b, f32x8 c) {
  // 8 args: (neg_a, A, neg_b, B, c_mod, C, reuse_a, reuse_b)
  return __builtin_amdgcn_wmma_f32_16x16x32_bf16(false, a, false, b,
                                                 (short)0, c, false, false);
}

__device__ __forceinline__ f32x8 zero8() {
  f32x8 z;
#pragma unroll
  for (int i = 0; i < 8; ++i) z[i] = 0.f;
  return z;
}

// Copy 16B global -> LDS. Async (ASYNCcnt-tracked) when available.
__device__ __forceinline__ void cp16_g2l(const bf16* g, bf16* l) {
#if USE_ASYNC_LDS
  __builtin_amdgcn_global_load_async_to_lds_b128(
      (AS1 i32x4*)(void*)g, (AS3 i32x4*)(void*)l, 0, 0);
#else
  *(bf16x8*)l = *(const bf16x8*)g;
#endif
}

template <int N>
__device__ __forceinline__ void wait_async() {
#if USE_ASYNC_LDS
  __builtin_amdgcn_s_wait_asynccnt(N);
#endif
}

// A-fragment of a 16x32 bf16 tile, row-major source. `p` = row base + k-offset.
// Element e -> K = hf*16 + e  (contiguous order, matches B fragments).
__device__ __forceinline__ bf16x16 load_a_frag(const bf16* p, int hf) {
  bf16x8 lo = *(const bf16x8*)(p + hf * 16);
  bf16x8 hi = *(const bf16x8*)(p + hf * 16 + 8);
  return __builtin_shufflevector(lo, hi, 0, 1, 2, 3, 4, 5, 6, 7,
                                 8, 9, 10, 11, 12, 13, 14, 15);
}

// ---------------------------------------------------------------------------
// Precision casts / weight transposes
// ---------------------------------------------------------------------------
__global__ void cast_bf16_kernel(const float* __restrict__ in,
                                 bf16* __restrict__ out, int n) {
  int i = blockIdx.x * blockDim.x + threadIdx.x;
  if (i < n) out[i] = (bf16)in[i];
}

// out[n*rows + k] = in[k*cols + n]  (weights [K,N] -> [N,K] bf16)
__global__ void transpose_cast_kernel(const float* __restrict__ in,
                                      bf16* __restrict__ out, int rows, int cols) {
  int i = blockIdx.x * blockDim.x + threadIdx.x;
  if (i < rows * cols) {
    int n = i / rows;
    int k = i % rows;
    out[i] = (bf16)in[(size_t)k * cols + n];
  }
}

// ---------------------------------------------------------------------------
// Shared GEMM core: each block = 4 waves = 128x64 tile; wave tile 32x64.
// B k-slab (64 n x 32 k bf16 = 4KB) double-buffered in LDS, async-staged and
// shared by all 4 waves; A streamed directly from global.
// smem: bf16[2][64][32] = 8KB.
// ---------------------------------------------------------------------------
__device__ __forceinline__ void gemm_tile_core(const bf16* __restrict__ A,
                                               const bf16* __restrict__ Bt,
                                               bf16* smem,
                                               int mbase, int nbase,
                                               f32x8 acc[2][4]) {
  const int tid  = threadIdx.x;
  const int lane = tid & 31;
  const int l16  = lane & 15;
  const int hf   = lane >> 4;
  const int w    = (tid >> 5) & 3;

  const bf16* arow0 = A + (size_t)(mbase + l16) * kC;
  const bf16* arow1 = arow0 + (size_t)16 * kC;

  // Stage one 64x32 B slab (k offset kb) into LDS buffer `buf`.
  // 256 16B chunks; each wave issues 2 async b128 ops (32 lanes x 16B each).
  auto stage = [&](int buf, int kb) {
#pragma unroll
    for (int j = 0; j < 2; ++j) {
      const int c  = w * 64 + j * 32 + lane;  // chunk 0..255
      const int nl = c >> 2;                  // slab row (n within tile)
      const int ke = (c & 3) * 8;             // k element offset
      cp16_g2l(Bt + (size_t)(nbase + nl) * kC + kb + ke,
               smem + buf * 2048 + c * 8);
    }
  };

  stage(0, 0);
  int buf = 0;
  for (int kb = 0; kb < kC; kb += 32) {
    if (kb + 32 < kC) {
      stage(buf ^ 1, kb + 32);
      wait_async<2>();   // current slab's 2 ops done; next slab in flight
    } else {
      wait_async<0>();
    }
    __syncthreads();     // all waves' slab writes visible

    bf16x16 a0 = load_a_frag(arow0 + kb, hf);
    bf16x16 a1 = load_a_frag(arow1 + kb, hf);
    const bf16* bbase = smem + buf * 2048 + hf * 16;
#pragma unroll
    for (int t = 0; t < 4; ++t) {
      // B frag: lane n = t*16+l16, element e -> k = hf*16 + e (contig 32B)
      bf16x16 bfr = *(const bf16x16*)(bbase + (t * 16 + l16) * 32);
      acc[0][t] = wmma_bf16(a0, bfr, acc[0][t]);
      acc[1][t] = wmma_bf16(a1, bfr, acc[1][t]);
    }
    __syncthreads();     // reads done before this buffer is overwritten
    buf ^= 1;
  }
}

// ---------------------------------------------------------------------------
// QKV GEMM: qkv = xb[M,C] @ w[C,3C] + bias.
// Epilogue scatters into q[B,H,T,D], k[B,H,T,D], vT[B,H,D,T] (bf16).
// ---------------------------------------------------------------------------
__global__ void qkv_gemm_kernel(const bf16* __restrict__ A,
                                const bf16* __restrict__ Bt,   // [N3, C]
                                const float* __restrict__ bias,
                                bf16* __restrict__ qh,
                                bf16* __restrict__ kh,
                                bf16* __restrict__ vth) {
  __shared__ bf16 smemB[4096];
  const int lane = threadIdx.x & 31;
  const int l16  = lane & 15;
  const int hf   = lane >> 4;
  const int w    = (threadIdx.x >> 5) & 3;
  const int mi = blockIdx.x % (kM / 128);
  const int ni = blockIdx.x / (kM / 128);
  const int mbase = mi * 128 + w * 32;
  const int nbase = ni * 64;

  f32x8 acc[2][4];
#pragma unroll
  for (int a = 0; a < 2; ++a)
#pragma unroll
    for (int b = 0; b < 4; ++b) acc[a][b] = zero8();

  gemm_tile_core(A, Bt, smemB, mbase, nbase, acc);

#pragma unroll
  for (int ft = 0; ft < 4; ++ft) {
    const int n    = nbase + ft * 16 + l16;
    const float bv = bias[n];
    const int sec  = n / kC;          // 0=q, 1=k, 2=v (uniform per frag)
    const int hidx = (n % kC) >> 6;   // head
    const int d    = n & 63;
#pragma unroll
    for (int mt = 0; mt < 2; ++mt) {
#pragma unroll
      for (int r = 0; r < 8; ++r) {
        const int m    = mbase + mt * 16 + hf * 8 + r;
        const int bidx = m >> 11;     // / 2048
        const int t    = m & 2047;
        const float v  = acc[mt][ft][r] + bv;
        const size_t bh = (size_t)(bidx * kH + hidx);
        if (sec == 0)      qh[(bh * kT + t) * kD + d]  = (bf16)v;
        else if (sec == 1) kh[(bh * kT + t) * kD + d]  = (bf16)v;
        else               vth[(bh * kD + d) * kT + t] = (bf16)v;
      }
    }
  }
}

// ---------------------------------------------------------------------------
// Flash attention: one wave per (b,h, 16-query block).
// Computes S^T = K_tile @ Q^T so softmax reductions stay intra-lane
// (lane = query column), then O^T = V^T_tile @ P^T.  Output -> yb[M,C] bf16.
// ---------------------------------------------------------------------------
__global__ void attn_kernel(const bf16* __restrict__ q,
                            const bf16* __restrict__ k,
                            const bf16* __restrict__ vt,
                            bf16* __restrict__ yb) {
  const int lane = threadIdx.x & 31;
  const int l16  = lane & 15;
  const int hf   = lane >> 4;
  const int wave = blockIdx.x * (blockDim.x >> 5) + (threadIdx.x >> 5);
  const int qtile = wave & (kT / 16 - 1);   // 0..127
  const int bh    = wave >> 7;              // 0..47
  const int qbase = qtile * 16;

  const bf16* qh_  = q  + (size_t)bh * kT * kD;
  const bf16* kh_  = k  + (size_t)bh * kT * kD;
  const bf16* vth_ = vt + (size_t)bh * kD * kT;

  // Q^T B-fragments (lane n = query qi; elements = head dim), kept all loop.
  const bf16* qrow = qh_ + (size_t)(qbase + l16) * kD + hf * 16;
  const bf16x16 qf0 = *(const bf16x16*)(qrow);
  const bf16x16 qf1 = *(const bf16x16*)(qrow + 32);

  const int qi = qbase + l16;               // this lane's query row
  float mrow = -__builtin_inff();
  float lrow = 0.f;
  f32x8 o[4];
#pragma unroll
  for (int dt = 0; dt < 4; ++dt) o[dt] = zero8();

  const float scale = 0.125f;               // 1/sqrt(64)
  const int kt_end = (qbase + 16 + 31) >> 5;

  for (int kt = 0; kt < kt_end; ++kt) {
    const int kbase = kt * 32;

    // S^T = K_tile(32xD) @ Q^T(Dx16): 4 WMMAs
    const bf16* kr0 = kh_ + (size_t)(kbase + l16) * kD;
    const bf16* kr1 = kr0 + (size_t)16 * kD;
    f32x8 s[2];
    s[0] = zero8();
    s[1] = zero8();
    s[0] = wmma_bf16(load_a_frag(kr0, hf), qf0, s[0]);
    s[0] = wmma_bf16(load_a_frag(kr0 + 32, hf), qf1, s[0]);
    s[1] = wmma_bf16(load_a_frag(kr1, hf), qf0, s[1]);
    s[1] = wmma_bf16(load_a_frag(kr1 + 32, hf), qf1, s[1]);

    // scale + causal mask; C-frag rows are key indices kj = kbase+mt*16+hf*8+r
    float cmax = -__builtin_inff();
#pragma unroll
    for (int mt = 0; mt < 2; ++mt) {
#pragma unroll
      for (int r = 0; r < 8; ++r) {
        const int kj = kbase + mt * 16 + hf * 8 + r;
        float sv = s[mt][r] * scale;
        sv = (kj <= qi) ? sv : -__builtin_inff();
        s[mt][r] = sv;
        cmax = fmaxf(cmax, sv);
      }
    }
    cmax = fmaxf(cmax, __shfl_xor(cmax, 16));

    const float newm = fmaxf(mrow, cmax);
    float rsum = 0.f;
#pragma unroll
    for (int mt = 0; mt < 2; ++mt) {
#pragma unroll
      for (int r = 0; r < 8; ++r) {
        const float p = __expf(s[mt][r] - newm);
        s[mt][r] = p;
        rsum += p;
      }
    }
    rsum += __shfl_xor(rsum, 16);

    const float alpha = __expf(mrow - newm);
    mrow = newm;
    lrow = lrow * alpha + rsum;
#pragma unroll
    for (int dt = 0; dt < 4; ++dt)
#pragma unroll
      for (int r = 0; r < 8; ++r) o[dt][r] *= alpha;

    // Build P^T B-fragment (lane n = qi, element e -> kj = kbase + hf*16 + e)
    // via a single lane-pair exchange instead of an LDS transpose.
    bf16x16 pb;
#pragma unroll
    for (int e = 0; e < 8; ++e) {
      const float own0 = s[0][e];
      const float oth0 = __shfl_xor(s[0][e], 16);
      const float own1 = s[1][e];
      const float oth1 = __shfl_xor(s[1][e], 16);
      pb[e]     = (bf16)(hf ? oth1 : own0);
      pb[e + 8] = (bf16)(hf ? own1 : oth0);
    }

    // O^T += V^T_tile(16x32) @ P^T(32x16): 4 WMMAs (one per 16 head dims)
#pragma unroll
    for (int dt = 0; dt < 4; ++dt) {
      const bf16* vr = vth_ + (size_t)(dt * 16 + l16) * kT + kbase;
      bf16x16 va = load_a_frag(vr, hf);
      o[dt] = wmma_bf16(va, pb, o[dt]);
    }
  }

  // Epilogue: O^T frag rows are head dims d = dt*16 + hf*8 + r (contiguous),
  // lane column is qi -> one 16B bf16 store per fragment per lane.
  const float inv = 1.0f / lrow;
  const int bidx = bh / kH;
  const int hidx = bh % kH;
  bf16* orow = yb + ((size_t)(bidx * kT + qi)) * kC + hidx * kD + hf * 8;
#pragma unroll
  for (int dt = 0; dt < 4; ++dt) {
    bf16x8 ov;
#pragma unroll
    for (int r = 0; r < 8; ++r) ov[r] = (bf16)(o[dt][r] * inv);
    *(bf16x8*)(orow + dt * 16) = ov;
  }
}

// ---------------------------------------------------------------------------
// Output projection GEMM: out = yb[M,C] @ w_proj[C,C] + b_proj (fp32 out).
// ---------------------------------------------------------------------------
__global__ void proj_gemm_kernel(const bf16* __restrict__ A,
                                 const bf16* __restrict__ Bt,  // [C, C]
                                 const float* __restrict__ bias,
                                 float* __restrict__ out) {
  __shared__ bf16 smemB[4096];
  const int lane = threadIdx.x & 31;
  const int l16  = lane & 15;
  const int hf   = lane >> 4;
  const int w    = (threadIdx.x >> 5) & 3;
  const int mi = blockIdx.x % (kM / 128);
  const int ni = blockIdx.x / (kM / 128);
  const int mbase = mi * 128 + w * 32;
  const int nbase = ni * 64;

  f32x8 acc[2][4];
#pragma unroll
  for (int a = 0; a < 2; ++a)
#pragma unroll
    for (int b = 0; b < 4; ++b) acc[a][b] = zero8();

  gemm_tile_core(A, Bt, smemB, mbase, nbase, acc);

#pragma unroll
  for (int ft = 0; ft < 4; ++ft) {
    const int n    = nbase + ft * 16 + l16;
    const float bv = bias[n];
#pragma unroll
    for (int mt = 0; mt < 2; ++mt) {
#pragma unroll
      for (int r = 0; r < 8; ++r) {
        const int m = mbase + mt * 16 + hf * 8 + r;
        out[(size_t)m * kC + n] = acc[mt][ft][r] + bv;
      }
    }
  }
}

// ---------------------------------------------------------------------------
// Host launcher
// ---------------------------------------------------------------------------
extern "C" void kernel_launch(void* const* d_in, const int* in_sizes, int n_in,
                              void* d_out, int out_size, void* d_ws, size_t ws_size,
                              hipStream_t stream) {
  const float* x      = (const float*)d_in[0];   // [B,T,C]
  const float* w_attn = (const float*)d_in[1];   // [C,3C]
  const float* b_attn = (const float*)d_in[2];   // [3C]
  const float* w_proj = (const float*)d_in[3];   // [C,C]
  const float* b_proj = (const float*)d_in[4];   // [C]
  float* out = (float*)d_out;                    // [B,T,C] fp32

  // Workspace layout (bf16, 256B-aligned regions), ~65 MB total.
  char* ws = (char*)d_ws;
  size_t off = 0;
  auto alloc = [&](size_t bytes) {
    void* p = ws + off;
    off += (bytes + 255) & ~(size_t)255;
    return p;
  };
  bf16* xb   = (bf16*)alloc((size_t)kM * kC * 2);      // x in bf16
  bf16* wabT = (bf16*)alloc((size_t)kN3 * kC * 2);     // w_attn^T
  bf16* wpT  = (bf16*)alloc((size_t)kC * kC * 2);      // w_proj^T
  bf16* qh   = (bf16*)alloc((size_t)kB * kH * kT * kD * 2);  // [B,H,T,D]
  bf16* kh   = (bf16*)alloc((size_t)kB * kH * kT * kD * 2);  // [B,H,T,D]
  bf16* vth  = (bf16*)alloc((size_t)kB * kH * kD * kT * 2);  // [B,H,D,T]
  bf16* yb   = (bf16*)alloc((size_t)kM * kC * 2);      // attention out, bf16
  (void)ws_size; (void)n_in; (void)in_sizes; (void)out_size;

  {
    int n = kM * kC;
    cast_bf16_kernel<<<(n + 255) / 256, 256, 0, stream>>>(x, xb, n);
  }
  {
    int n = kC * kN3;
    transpose_cast_kernel<<<(n + 255) / 256, 256, 0, stream>>>(w_attn, wabT, kC, kN3);
  }
  {
    int n = kC * kC;
    transpose_cast_kernel<<<(n + 255) / 256, 256, 0, stream>>>(w_proj, wpT, kC, kC);
  }
  {
    // blocks = (M/128) * (3C/64) = 64*36 = 2304, 128 threads (4 waves)
    int blocks = (kM / 128) * (kN3 / 64);
    qkv_gemm_kernel<<<blocks, 128, 0, stream>>>(xb, wabT, b_attn, qh, kh, vth);
  }
  {
    // waves = B*H*(T/16) = 6144, 4 independent waves per block
    int waves = kB * kH * (kT / 16);
    attn_kernel<<<waves / 4, 128, 0, stream>>>(qh, kh, vth, yb);
  }
  {
    // blocks = (M/128) * (C/64) = 64*12 = 768
    int blocks = (kM / 128) * (kC / 64);
    proj_gemm_kernel<<<blocks, 128, 0, stream>>>(yb, wpT, b_proj, out);
  }
}
